// SemanticSlotEncoder_34419867910743
// MI455X (gfx1250) — matrix operand order repaired and checked by reading, not compiled, not run on hardware
//
#include <hip/hip_runtime.h>
#include <hip/hip_bf16.h>

// ---------------------------------------------------------------------------
// SemanticSlotEncoder for MI455X (gfx1250, wave32, WMMA).
// All dense algebra runs through v_wmma_f32_16x16x32_f16 (f16 in, f32 acc).
// GEMM: fragment-layout LDS tiles (contiguous 32B per lane -> ds_load_b128),
// register-prefetch + double-buffered LDS software pipeline so global loads
// for tile k+1 overlap the WMMAs of tile k.
// ---------------------------------------------------------------------------

typedef __attribute__((ext_vector_type(16))) _Float16 v16h;
typedef __attribute__((ext_vector_type(2)))  _Float16 v2h;
typedef __attribute__((ext_vector_type(8)))  float    v8f;

#define CB 64      // batch
#define CH 128     // height
#define CW 128     // width
#define CPS 8      // patch size
#define CG 16      // grid
#define CN 256     // tokens = G*G
#define CE 256     // embed
#define CHEADS 4
#define CHD 64
#define CFD 320
#define CNS 8
#define CIT 3
#define CBN (CB*CN)          // 16384 rows
#define CKIM 768             // im2col K = 12*8*8

// ---------------------------------------------------------------------------
// Generic WMMA GEMM:  C[M,N] = act(alpha * A[M,K] @ B + bias) (+ Cres)
//  BT=true : B element (k,n) = Bsrc[n*ldb + k]   (i.e. x @ W^T, W row-major [N,K])
//  BT=false: B element (k,n) = Bsrc[k*ldb + n]
//  ACT: 0 = none, 1 = exact GELU
// Batched over gridDim.z with element strides sA/sB/sC/sRes.
// Block: 128 threads = 4 waves; 64x64 output tile; wave w -> rows w*16..w*16+15.
// Requires K % 32 == 0 (true for every call site: 768,256,512,64,320,640,960).
// ---------------------------------------------------------------------------
template<bool BT, int ACT>
__global__ __launch_bounds__(128)
void gemm_f16_wmma(const float* __restrict__ A, const float* __restrict__ Bsrc,
                   const float* __restrict__ bias, const float* __restrict__ Cres,
                   float* __restrict__ C,
                   int M, int N, int K, int lda, int ldb, int ldc,
                   long long sA, long long sB, long long sC, long long sRes,
                   float alpha)
{
    // Double-buffered fragment-layout LDS tiles (16 contiguous halves / lane).
    __shared__ __align__(32) _Float16 lsA[2][64 * 32];
    __shared__ __align__(32) _Float16 lsB[2][64 * 32];

    const int tid  = threadIdx.x;
    const int wave = tid >> 5;          // 0..3 (wave32)
    const int lane = tid & 31;
    const int bm = blockIdx.y * 64;
    const int bn = blockIdx.x * 64;
    const long long bz = blockIdx.z;

    A    += bz * sA;
    Bsrc += bz * sB;
    C    += bz * sC;
    const float* Rp = Cres ? (Cres + bz * sRes) : nullptr;

    // Block-uniform raggedness flags (scalar branches, no exec juggling).
    const bool fullA = (bm + 64 <= M);
    const bool fullB = (bn + 64 <= N);

    // Register prefetch file: one 64x32 A tile + one 32x64 B tile per step.
    float2 ra[8], rb[8];

    // ---- load phase: 16 global_load_b64 into registers, no waits here ----
    auto loadTiles = [&](int k0) {
        if (fullA) {
            #pragma unroll
            for (int i2 = 0; i2 < 8; ++i2) {
                int p = tid + (i2 << 7);
                int r = p >> 4, c = (p & 15) << 1;
                const float* ap = A + (long long)(bm + r) * lda + (k0 + c);
                ra[i2].x = ap[0]; ra[i2].y = ap[1];
            }
        } else {
            #pragma unroll
            for (int i2 = 0; i2 < 8; ++i2) {
                int p = tid + (i2 << 7);
                int r = p >> 4, c = (p & 15) << 1;
                int gr = bm + r;
                ra[i2].x = 0.0f; ra[i2].y = 0.0f;
                if (gr < M) {
                    const float* ap = A + (long long)gr * lda + (k0 + c);
                    ra[i2].x = ap[0]; ra[i2].y = ap[1];
                }
            }
        }
        if (fullB) {
            #pragma unroll
            for (int i2 = 0; i2 < 8; ++i2) {
                int p = tid + (i2 << 7);
                int kk = (p >> 6) << 1, nn = p & 63;
                int gk = k0 + kk, gn = bn + nn;
                if (BT) {
                    const float* bp = Bsrc + (long long)gn * ldb + gk;
                    rb[i2].x = bp[0]; rb[i2].y = bp[1];
                } else {
                    rb[i2].x = Bsrc[(long long)gk * ldb + gn];
                    rb[i2].y = Bsrc[(long long)(gk + 1) * ldb + gn];
                }
            }
        } else {
            #pragma unroll
            for (int i2 = 0; i2 < 8; ++i2) {
                int p = tid + (i2 << 7);
                int kk = (p >> 6) << 1, nn = p & 63;
                int gk = k0 + kk, gn = bn + nn;
                rb[i2].x = 0.0f; rb[i2].y = 0.0f;
                if (gn < N) {
                    if (BT) {
                        const float* bp = Bsrc + (long long)gn * ldb + gk;
                        rb[i2].x = bp[0]; rb[i2].y = bp[1];
                    } else {
                        rb[i2].x = Bsrc[(long long)gk * ldb + gn];
                        rb[i2].y = Bsrc[(long long)(gk + 1) * ldb + gn];
                    }
                }
            }
        }
    };

    // ---- commit phase: v_cvt_pk_f16_f32 + ds_store_b32 into fragment layout
    // A mapping: lane = 16*((c>>3)&1) + (r&15); j = (c&7) + 8*(c>=16)
    // B mapping: lane = 16*((kk>>3)&1) + (nn&15); j = (kk&7) + 8*(kk>=16); t = nn>>4
    auto storeTiles = [&](int buf) {
        #pragma unroll
        for (int i2 = 0; i2 < 8; ++i2) {
            int p = tid + (i2 << 7);
            int r = p >> 4, c = (p & 15) << 1;
            int ln = (((c >> 3) & 1) << 4) + (r & 15);
            int j  = (c & 7) + ((c >> 4) << 3);
            v2h pk = { (_Float16)ra[i2].x, (_Float16)ra[i2].y };
            *(v2h*)&lsA[buf][((((r >> 4) << 5) + ln) << 4) + j] = pk;
        }
        #pragma unroll
        for (int i2 = 0; i2 < 8; ++i2) {
            int p = tid + (i2 << 7);
            int kk = (p >> 6) << 1, nn = p & 63;
            int t  = nn >> 4;
            int ln = (((kk >> 3) & 1) << 4) + (nn & 15);
            int j  = (kk & 7) + ((kk >> 4) << 3);
            v2h pk = { (_Float16)rb[i2].x, (_Float16)rb[i2].y };
            *(v2h*)&lsB[buf][(((t << 5) + ln) << 4) + j] = pk;
        }
    };

    v8f acc[4] = {};                    // 4 N-subtiles of 16 cols each

    // Prologue: stage tile 0.
    loadTiles(0);
    storeTiles(0);
    __syncthreads();

    const int nk = K >> 5;
    for (int ki = 0; ki < nk; ++ki) {
        const int cur = ki & 1;
        const bool more = (ki + 1) < nk;
        if (more) loadTiles((ki + 1) << 5);   // global loads overlap WMMAs below

        const v16h af = *(const v16h*)&lsA[cur][((wave << 5) + lane) << 4];
        #pragma unroll
        for (int t = 0; t < 4; ++t) {
            const v16h bf = *(const v16h*)&lsB[cur][((t << 5) + lane) << 4];
            acc[t] = __builtin_amdgcn_wmma_f32_16x16x32_f16(
                         false, af, false, bf, (short)0, acc[t], false, false);
        }
        __syncthreads();
        if (more) {
            storeTiles(cur ^ 1);
            __syncthreads();
        }
    }

    // Epilogue. f32 C/D layout: VGPR r holds (row = r + 8*(lane>=16), col = lane&15)
    const int colInTile = lane & 15;
    const int rowHi = (lane >> 4) * 8;
    #pragma unroll
    for (int t = 0; t < 4; ++t) {
        int gn = bn + t * 16 + colInTile;
        if (gn >= N) continue;
        float bv = bias ? bias[gn] : 0.0f;
        #pragma unroll
        for (int r = 0; r < 8; ++r) {
            int gm = bm + wave * 16 + rowHi + r;
            if (gm >= M) continue;
            float v = acc[t][r] * alpha + bv;
            if (ACT == 1) v = 0.5f * v * (1.0f + erff(v * 0.70710678118f));
            if (Rp) v += Rp[(long long)gm * ldc + gn];
            C[(long long)gm * ldc + gn] = v;
        }
    }
}

// ---------------------------------------------------------------------------
// im2col with fused one-hot (V=10) + Sobel-x + Sobel-y (edge-clamped).
// Acol[(b*256 + gy*16 + gx)][c*64 + py*8 + px], matching conv_w [E][12][8][8].
// ---------------------------------------------------------------------------
__global__ void im2col_kernel(const float* __restrict__ state, float* __restrict__ Acol)
{
    long long idx = (long long)blockIdx.x * blockDim.x + threadIdx.x;
    const long long total = (long long)CBN * CKIM;
    if (idx >= total) return;
    int col = (int)(idx % CKIM);
    long long row = idx / CKIM;
    int b = (int)(row / CN);
    int n = (int)(row % CN);
    int gy = n >> 4, gx = n & 15;
    int c = col >> 6;
    int py = (col >> 3) & 7, px = col & 7;
    int y = gy * CPS + py, x = gx * CPS + px;
    const float* S = state + (long long)b * CH * CW;
    float v;
    if (c < 10) {
        v = ((int)S[y * CW + x] == c) ? 1.0f : 0.0f;
    } else {
        auto at = [&](int yy, int xx) {
            yy = yy < 0 ? 0 : (yy > CH - 1 ? CH - 1 : yy);
            xx = xx < 0 ? 0 : (xx > CW - 1 ? CW - 1 : xx);
            return S[yy * CW + xx];
        };
        if (c == 10)  // sobel x
            v = -at(y-1,x-1) + at(y-1,x+1) - 2.0f*at(y,x-1) + 2.0f*at(y,x+1)
                - at(y+1,x-1) + at(y+1,x+1);
        else          // sobel y
            v = -at(y-1,x-1) - 2.0f*at(y-1,x) - at(y-1,x+1)
                + at(y+1,x-1) + 2.0f*at(y+1,x) + at(y+1,x+1);
    }
    Acol[idx] = v;
}

// ---------------------------------------------------------------------------
// LayerNorm: one wave (32 lanes) per row, shuffle reductions.
// ---------------------------------------------------------------------------
__global__ void layernorm_kernel(const float* __restrict__ x,
                                 const float* __restrict__ g, const float* __restrict__ b,
                                 float* __restrict__ y, int rows, int D)
{
    int row  = blockIdx.x * (blockDim.x >> 5) + (threadIdx.x >> 5);
    int lane = threadIdx.x & 31;
    if (row >= rows) return;
    const float* xr = x + (long long)row * D;
    float s = 0.0f;
    for (int i = lane; i < D; i += 32) s += xr[i];
    for (int o = 16; o > 0; o >>= 1) s += __shfl_xor(s, o, 32);
    float mean = s / (float)D;
    float v = 0.0f;
    for (int i = lane; i < D; i += 32) { float d = xr[i] - mean; v += d * d; }
    for (int o = 16; o > 0; o >>= 1) v += __shfl_xor(v, o, 32);
    float rstd = rsqrtf(v / (float)D + 1e-5f);
    float* yr = y + (long long)row * D;
    for (int i = lane; i < D; i += 32) yr[i] = (xr[i] - mean) * rstd * g[i] + b[i];
}

// ---------------------------------------------------------------------------
// Row softmax in place: one wave per row.
// ---------------------------------------------------------------------------
__global__ void softmax_kernel(float* __restrict__ x, long long rows, int D)
{
    long long row = (long long)blockIdx.x * (blockDim.x >> 5) + (threadIdx.x >> 5);
    int lane = threadIdx.x & 31;
    if (row >= rows) return;
    float* xr = x + row * D;
    float mx = -3.4e38f;
    for (int i = lane; i < D; i += 32) mx = fmaxf(mx, xr[i]);
    for (int o = 16; o > 0; o >>= 1) mx = fmaxf(mx, __shfl_xor(mx, o, 32));
    float s = 0.0f;
    for (int i = lane; i < D; i += 32) { float e = expf(xr[i] - mx); xr[i] = e; s += e; }
    for (int o = 16; o > 0; o >>= 1) s += __shfl_xor(s, o, 32);
    float inv = 1.0f / s;
    for (int i = lane; i < D; i += 32) xr[i] *= inv;
}

// ---------------------------------------------------------------------------
// RoPE on q,k + split/transpose qkv[B,N,3,H,HD] -> qT/kT/vT [B*H, N, HD].
// ---------------------------------------------------------------------------
__global__ void rope_split_kernel(const float* __restrict__ qkv,
                                  float* __restrict__ qT, float* __restrict__ kT,
                                  float* __restrict__ vT)
{
    int idx = blockIdx.x * blockDim.x + threadIdx.x;
    if (idx >= CB * CHEADS * CN * CHD) return;
    int d = idx & 63;
    int n = (idx >> 6) & 255;
    int h = (idx >> 14) & 3;
    int b = idx >> 16;
    long long base = (((long long)b * CN + n) * 3) * (CHEADS * CHD) + h * CHD;
    float q = qkv[base + d];
    float k = qkv[base + CHEADS * CHD + d];
    float v = qkv[base + 2 * CHEADS * CHD + d];
    int gy = n >> 4, gx = n & 15;
    const int half = CHD / 2;                  // 32
    int dd = (d < half) ? d : d - half;
    int g  = (d < half) ? gy : gx;
    int jj = (dd < 16) ? dd : dd - 16;
    float inv = powf(10000.0f, -((float)(2 * jj) / 32.0f));
    float p = (float)g * inv;
    float e = (dd < 16) ? sinf(p) : cosf(p);   // the "emb" value
    float ce = cosf(e), se = sinf(e);
    int dr = (d < half) ? (2 * d + 1) : (2 * (d - half));
    float sgn = (d < half) ? -1.0f : 1.0f;
    float qr = qkv[base + dr];
    float kr = qkv[base + CHEADS * CHD + dr];
    long long o = (((long long)b * CHEADS + h) * CN + n) * CHD + d;
    qT[o] = q * ce + sgn * qr * se;
    kT[o] = k * ce + sgn * kr * se;
    vT[o] = v;
}

// aoH [B*H, N, HD] -> ao [B, N, H*HD]
__global__ void permute_ao_kernel(const float* __restrict__ aoH, float* __restrict__ ao)
{
    int idx = blockIdx.x * blockDim.x + threadIdx.x;
    if (idx >= CB * CN * CE) return;
    int e = idx & 255;
    int n = (idx >> 8) & 255;
    int b = idx >> 16;
    int h = e >> 6, d = e & 63;
    ao[idx] = aoH[(((long long)b * CHEADS + h) * CN + n) * CHD + d];
}

// ---------------------------------------------------------------------------
// Slot attention: softmax over s per (b,n) -> sattn; then an = sattn+1e-8
// normalized over n, written transposed anT[B, NS, N]. One block per batch.
// ---------------------------------------------------------------------------
__global__ __launch_bounds__(256)
void slot_attn_kernel(const float* __restrict__ logits,
                      float* __restrict__ sattn, float* __restrict__ anT)
{
    int b = blockIdx.x, n = threadIdx.x;
    __shared__ float red[256];
    __shared__ float colsum[CNS];
    const float* L = logits + ((long long)b * CN + n) * CNS;
    float a[CNS];
    float mx = -3.4e38f;
    #pragma unroll
    for (int s = 0; s < CNS; ++s) { a[s] = L[s]; mx = fmaxf(mx, a[s]); }
    float sum = 0.0f;
    #pragma unroll
    for (int s = 0; s < CNS; ++s) { a[s] = expf(a[s] - mx); sum += a[s]; }
    float inv = 1.0f / sum;
    float* So = sattn + ((long long)b * CN + n) * CNS;
    #pragma unroll
    for (int s = 0; s < CNS; ++s) { a[s] *= inv; So[s] = a[s]; a[s] += 1e-8f; }
    for (int s = 0; s < CNS; ++s) {
        red[n] = a[s];
        __syncthreads();
        for (int off = 128; off > 0; off >>= 1) {
            if (n < off) red[n] += red[n + off];
            __syncthreads();
        }
        if (n == 0) colsum[s] = red[0];
        __syncthreads();
    }
    float* An = anT + (long long)b * CNS * CN;
    #pragma unroll
    for (int s = 0; s < CNS; ++s) An[s * CN + n] = a[s] / colsum[s];
}

// GRU gates: slots_new = (1-z)*tanh(i_n + r*h_n) + z*h
__global__ void gru_gate_kernel(const float* __restrict__ gi, const float* __restrict__ gh,
                                const float* __restrict__ hprev, float* __restrict__ hnew,
                                int rows, int fd)
{
    int idx = blockIdx.x * blockDim.x + threadIdx.x;
    if (idx >= rows * fd) return;
    int row = idx / fd, f = idx % fd;
    const float* gir = gi + (long long)row * 3 * fd;
    const float* ghr = gh + (long long)row * 3 * fd;
    float r  = 1.0f / (1.0f + expf(-(gir[f] + ghr[f])));
    float z  = 1.0f / (1.0f + expf(-(gir[fd + f] + ghr[fd + f])));
    float nn = tanhf(gir[2 * fd + f] + r * ghr[2 * fd + f]);
    float h  = hprev[idx];
    hnew[idx] = (1.0f - z) * nn + z * h;
}

// Pack outputs: slots flat, then masks[b][s][g][g] = sattn[b][n][s]
__global__ void output_kernel(const float* __restrict__ slots,
                              const float* __restrict__ sattn, float* __restrict__ out)
{
    const int SLOT_N = CB * CNS * CFD;   // 163840
    const int MASK_N = CB * CNS * CN;    // 131072
    int idx = blockIdx.x * blockDim.x + threadIdx.x;
    if (idx < SLOT_N) {
        out[idx] = slots[idx];
    } else if (idx < SLOT_N + MASK_N) {
        int i = idx - SLOT_N;
        int n = i % CN;
        int s = (i / CN) % CNS;
        int b = i / (CN * CNS);
        out[idx] = sattn[((long long)b * CN + n) * CNS + s];
    }
}

// ---------------------------------------------------------------------------
// Host side
// ---------------------------------------------------------------------------
static void gemm(hipStream_t st, bool bt, bool gelu,
                 const float* A, const float* Bsrc, const float* bias,
                 const float* Cres, float* C,
                 int M, int N, int K, int lda, int ldb, int ldc,
                 long long sA, long long sB, long long sC, long long sRes,
                 int batch, float alpha)
{
    dim3 g((N + 63) / 64, (M + 63) / 64, batch), blk(128);
    if (bt && gelu)
        gemm_f16_wmma<true, 1><<<g, blk, 0, st>>>(A, Bsrc, bias, Cres, C, M, N, K,
                                                  lda, ldb, ldc, sA, sB, sC, sRes, alpha);
    else if (bt)
        gemm_f16_wmma<true, 0><<<g, blk, 0, st>>>(A, Bsrc, bias, Cres, C, M, N, K,
                                                  lda, ldb, ldc, sA, sB, sC, sRes, alpha);
    else if (gelu)
        gemm_f16_wmma<false, 1><<<g, blk, 0, st>>>(A, Bsrc, bias, Cres, C, M, N, K,
                                                   lda, ldb, ldc, sA, sB, sC, sRes, alpha);
    else
        gemm_f16_wmma<false, 0><<<g, blk, 0, st>>>(A, Bsrc, bias, Cres, C, M, N, K,
                                                   lda, ldb, ldc, sA, sB, sC, sRes, alpha);
}

extern "C" void kernel_launch(void* const* d_in, const int* in_sizes, int n_in,
                              void* d_out, int out_size, void* d_ws, size_t ws_size,
                              hipStream_t stream)
{
    const float* state    = (const float*)d_in[0];
    const float* conv_w   = (const float*)d_in[1];
    const float* conv_b   = (const float*)d_in[2];
    const float* mlp_ln_g = (const float*)d_in[3];
    const float* mlp_ln_b = (const float*)d_in[4];
    const float* mlp_w1   = (const float*)d_in[5];
    const float* mlp_b1   = (const float*)d_in[6];
    const float* mlp_w2   = (const float*)d_in[7];
    const float* mlp_b2   = (const float*)d_in[8];
    const float* qkv_w    = (const float*)d_in[9];
    const float* qkv_b    = (const float*)d_in[10];
    const float* proj_w   = (const float*)d_in[11];
    const float* proj_b   = (const float*)d_in[12];
    const float* ni_g     = (const float*)d_in[13];
    const float* ni_b     = (const float*)d_in[14];
    const float* ns_g     = (const float*)d_in[15];
    const float* ns_b     = (const float*)d_in[16];
    const float* nm_g     = (const float*)d_in[17];
    const float* nm_b     = (const float*)d_in[18];
    const float* q_w      = (const float*)d_in[19];
    const float* k_w      = (const float*)d_in[20];
    const float* v_w      = (const float*)d_in[21];
    const float* gru_wih  = (const float*)d_in[22];
    const float* gru_whh  = (const float*)d_in[23];
    const float* gru_bih  = (const float*)d_in[24];
    const float* gru_bhh  = (const float*)d_in[25];
    const float* smlp_w1  = (const float*)d_in[26];
    const float* smlp_b1  = (const float*)d_in[27];
    const float* smlp_w2  = (const float*)d_in[28];
    const float* smlp_b2  = (const float*)d_in[29];

    float* W = (float*)d_ws;

    // ---- workspace layout (floats), with region reuse ----
    const long long R0      = 0;                       // 16,777,216 f: im2col -> scores -> k_in/v_in
    const long long O_x     = 16777216;                // x (post-MLP token features)
    const long long O_tmpA  = 20971520;                // h/h1 -> qkv -> aoH/ao
    const long long O_R3    = 33554432;                // qT/kT/vT -> x2/xn
    const long long O_small = 46137344;

    float* Acol  = W + R0;
    float* scores= W + R0;
    float* k_in  = W + R0;
    float* v_in  = W + R0 + 5242880;
    float* xb    = W + O_x;
    float* hb    = W + O_tmpA;
    float* h1b   = W + O_tmpA + 4194304;
    float* qkvb  = W + O_tmpA;
    float* aoH   = W + O_tmpA;
    float* aob   = W + O_tmpA + 4194304;
    float* qT    = W + O_R3;
    float* kT    = W + O_R3 + 4194304;
    float* vT    = W + O_R3 + 8388608;
    float* x2    = W + O_R3;               // reuse qT (dead)
    float* xn    = W + O_R3 + 4194304;     // reuse kT (dead)

    long long o = O_small;
    float* S      = W + o; o += 163840;    // slots
    float* S2     = W + o; o += 163840;
    float* sln    = W + o; o += 163840;
    float* q_s    = W + o; o += 163840;
    float* logits = W + o; o += 131072;
    float* sattn  = W + o; o += 131072;
    float* anT    = W + o; o += 131072;
    float* upd    = W + o; o += 163840;
    float* gi     = W + o; o += 491520;
    float* gh     = W + o; o += 491520;
    float* mb     = W + o; o += 163840;
    float* m1b    = W + o; o += 327680;

    const dim3 eb(256);
    auto eg = [](long long total) { return dim3((unsigned)((total + 255) / 256)); };

    // 1) im2col (onehot + sobel)
    im2col_kernel<<<eg((long long)CBN * CKIM), eb, 0, stream>>>(state, Acol);

    // 2) patch embed: x = Acol @ conv_w^T + conv_b     [16384 x 256], K=768
    gemm(stream, true, false, Acol, conv_w, conv_b, nullptr, xb,
         CBN, CE, CKIM, CKIM, CKIM, CE, 0, 0, 0, 0, 1, 1.0f);

    // 3) MLP: h = LN(x); h1 = gelu(h@w1^T+b1); x = h1@w2^T+b2
    layernorm_kernel<<<dim3(CBN / 4), dim3(128), 0, stream>>>(xb, mlp_ln_g, mlp_ln_b, hb, CBN, CE);
    gemm(stream, true, true,  hb,  mlp_w1, mlp_b1, nullptr, h1b,
         CBN, 2 * CE, CE, CE, CE, 2 * CE, 0, 0, 0, 0, 1, 1.0f);
    gemm(stream, true, false, h1b, mlp_w2, mlp_b2, nullptr, xb,
         CBN, CE, 2 * CE, 2 * CE, 2 * CE, CE, 0, 0, 0, 0, 1, 1.0f);

    // 4) QKV + RoPE + head transpose
    gemm(stream, true, false, xb, qkv_w, qkv_b, nullptr, qkvb,
         CBN, 3 * CE, CE, CE, CE, 3 * CE, 0, 0, 0, 0, 1, 1.0f);
    rope_split_kernel<<<eg((long long)CB * CHEADS * CN * CHD), eb, 0, stream>>>(qkvb, qT, kT, vT);

    // 5) attention: scores = softmax(q@k^T / 8); aoH = scores@v   (batch = B*H = 256)
    gemm(stream, true, false, qT, kT, nullptr, nullptr, scores,
         CN, CN, CHD, CHD, CHD, CN,
         (long long)CN * CHD, (long long)CN * CHD, (long long)CN * CN, 0,
         CB * CHEADS, 0.125f);
    softmax_kernel<<<dim3((unsigned)((long long)CB * CHEADS * CN / 4)), dim3(128), 0, stream>>>(
        scores, (long long)CB * CHEADS * CN, CN);
    gemm(stream, false, false, scores, vT, nullptr, nullptr, aoH,
         CN, CHD, CN, CN, CHD, CHD,
         (long long)CN * CN, (long long)CN * CHD, (long long)CN * CHD, 0,
         CB * CHEADS, 1.0f);
    permute_ao_kernel<<<eg((long long)CB * CN * CE), eb, 0, stream>>>(aoH, aob);

    // 6) proj + residual; LN(ni); k_in / v_in projections (E -> FD)
    gemm(stream, true, false, aob, proj_w, proj_b, xb, x2,
         CBN, CE, CE, CE, CE, CE, 0, 0, 0, 0, 1, 1.0f);
    layernorm_kernel<<<dim3(CBN / 4), dim3(128), 0, stream>>>(x2, ni_g, ni_b, xn, CBN, CE);
    gemm(stream, true, false, xn, k_w, nullptr, nullptr, k_in,
         CBN, CFD, CE, CE, CE, CFD, 0, 0, 0, 0, 1, 1.0f);
    gemm(stream, true, false, xn, v_w, nullptr, nullptr, v_in,
         CBN, CFD, CE, CE, CE, CFD, 0, 0, 0, 0, 1, 1.0f);

    // 7) slot attention, 3 iterations; slots start at zero
    hipMemsetAsync(S, 0, (size_t)CB * CNS * CFD * sizeof(float), stream);
    const int SR = CB * CNS;               // 512 slot rows
    const float logit_scale = 0.055901699f; // FD^-0.5 (TEMP = 1)
    for (int it = 0; it < CIT; ++it) {
        layernorm_kernel<<<dim3(SR / 4), dim3(128), 0, stream>>>(S, ns_g, ns_b, sln, SR, CFD);
        gemm(stream, true, false, sln, q_w, nullptr, nullptr, q_s,
             SR, CFD, CFD, CFD, CFD, CFD, 0, 0, 0, 0, 1, 1.0f);
        // logits[b] = k_in[b] @ q_s[b]^T * scale   [256 x 8], K=320, batch=B
        gemm(stream, true, false, k_in, q_s, nullptr, nullptr, logits,
             CN, CNS, CFD, CFD, CFD, CNS,
             (long long)CN * CFD, (long long)CNS * CFD, (long long)CN * CNS, 0,
             CB, logit_scale);
        slot_attn_kernel<<<dim3(CB), dim3(CN), 0, stream>>>(logits, sattn, anT);
        // updates[b] = anT[b] @ v_in[b]   [8 x 320], K=256, batch=B
        gemm(stream, false, false, anT, v_in, nullptr, nullptr, upd,
             CNS, CFD, CN, CN, CFD, CFD,
             (long long)CNS * CN, (long long)CN * CFD, (long long)CNS * CFD, 0,
             CB, 1.0f);
        // GRU
        gemm(stream, true, false, upd, gru_wih, gru_bih, nullptr, gi,
             SR, 3 * CFD, CFD, CFD, CFD, 3 * CFD, 0, 0, 0, 0, 1, 1.0f);
        gemm(stream, true, false, S, gru_whh, gru_bhh, nullptr, gh,
             SR, 3 * CFD, CFD, CFD, CFD, 3 * CFD, 0, 0, 0, 0, 1, 1.0f);
        gru_gate_kernel<<<eg((long long)SR * CFD), eb, 0, stream>>>(gi, gh, S, S2, SR, CFD);
        // slot MLP with residual
        layernorm_kernel<<<dim3(SR / 4), dim3(128), 0, stream>>>(S2, nm_g, nm_b, mb, SR, CFD);
        gemm(stream, true, true,  mb,  smlp_w1, smlp_b1, nullptr, m1b,
             SR, 2 * CFD, CFD, CFD, CFD, 2 * CFD, 0, 0, 0, 0, 1, 1.0f);
        gemm(stream, true, false, m1b, smlp_w2, smlp_b2, S2, S,
             SR, CFD, 2 * CFD, 2 * CFD, 2 * CFD, CFD, 0, 0, 0, 0, 1, 1.0f);
    }

    // 8) pack outputs: slots then masks (sattn transposed)
    output_kernel<<<eg((long long)CB * CNS * (CFD + CN)), eb, 0, stream>>>(S, sattn, (float*)d_out);
}
